// BLTTransformerLayer_52467320488185
// MI455X (gfx1250) — compile-verified, matrix operand order checked by reading, not run on hardware
//
#include <hip/hip_runtime.h>

// ---------------------------------------------------------------------------
// BLT transformer layer forward for MI455X (gfx1250), wave32 + WMMA bf16.
// Matmuls: bf16 x bf16 -> fp32 accumulate via v_wmma_f32_16x16x32_bf16.
// bf16 tile staging to LDS uses GLOBAL_LOAD_ASYNC_TO_LDS_B128 (ASYNCcnt).
// ---------------------------------------------------------------------------

typedef __attribute__((ext_vector_type(16))) __bf16 v16bf;
typedef __attribute__((ext_vector_type(8)))  float  v8f;

#define DEV __device__ __forceinline__

static constexpr int   Bb   = 2;
static constexpr int   S    = 2048;
static constexpr int   D    = 2048;
static constexpr int   H    = 16;
static constexpr int   KVH  = 4;
static constexpr int   HD   = 128;
static constexpr int   Ii   = 8192;
static constexpr int   M_ROWS = Bb * S;            // 4096
static constexpr float SCALE  = 0.08838834764831845f;  // 128^-0.5
static constexpr float NEG    = -1e9f;
static constexpr float EPS    = 1e-6f;

// fp32 -> bf16 via hardware convert (v_cvt_*bf16_f32), RNE.
DEV unsigned short f2bf(float f) {
  union { __bf16 b; unsigned short u; } x;
  x.b = (__bf16)f;
  return x.u;
}
DEV unsigned pack2(float a, float b) {
  return (unsigned)f2bf(a) | ((unsigned)f2bf(b) << 16);
}

union FragU { uint4 q[2]; v16bf v; };

// Load a 16xK=32 bf16 WMMA fragment (A layout; dense B layout mirrors A with
// N rows <-> M rows). `tile` points at the first row; rows are `ld` bf16
// elements apart, K-contiguous.
// Lane L (0-15): row L, K = k0+{0..7} and k0+{16..23}; lane L+16: +8.
DEV v16bf ld_frag(const unsigned short* tile, int row0, int k0, int ld, int lane) {
  const int r  = row0 + (lane & 15);
  const int kk = k0 + ((lane >> 4) << 3);
  const unsigned short* p = tile + (size_t)r * ld + kk;
  FragU u;
  u.q[0] = *(const uint4*)(p);
  u.q[1] = *(const uint4*)(p + 16);
  return u.v;
}

DEV v8f wmma_bf16(v16bf a, v16bf b, v8f c) {
  return __builtin_amdgcn_wmma_f32_16x16x32_bf16(false, a, false, b,
                                                 (short)0, c, false, false);
}

// 32-bit LDS byte offset of a __shared__ object (addrspace(3) pointers are
// 32-bit offsets from the wave's LDS base, which is what async VDST wants).
DEV unsigned lds_addr32(const void* p) {
  return (unsigned)(unsigned long long)
      (__attribute__((address_space(3))) const void*)p;
}

// Async global -> LDS 16-byte copy (per active lane); tracked by ASYNCcnt.
DEV void async_cp_b128(unsigned lds_off, const void* gaddr) {
  asm volatile("global_load_async_to_lds_b128 %0, %1, off"
               :: "v"(lds_off), "v"((unsigned long long)(size_t)gaddr)
               : "memory");
}
DEV void async_wait0() {
  asm volatile("s_wait_asynccnt 0" ::: "memory");
}

// ---------------------------------------------------------------------------
// RMSNorm: fp32 row (2048) -> bf16 row.  One block (256 thr) per row.
// ---------------------------------------------------------------------------
__global__ void rmsnorm_bf16_kernel(const float* __restrict__ x,
                                    const float* __restrict__ w,
                                    unsigned short* __restrict__ out) {
  __shared__ float red[8];
  const int row = blockIdx.x;
  const int tid = threadIdx.x;
  const size_t base = (size_t)row * D + tid * 8;
  float4 a = *(const float4*)(x + base);
  float4 b = *(const float4*)(x + base + 4);
  float ss = a.x*a.x + a.y*a.y + a.z*a.z + a.w*a.w
           + b.x*b.x + b.y*b.y + b.z*b.z + b.w*b.w;
  #pragma unroll
  for (int m = 16; m >= 1; m >>= 1) ss += __shfl_xor(ss, m, 32);
  if ((tid & 31) == 0) red[tid >> 5] = ss;
  __syncthreads();
  float tot = 0.f;
  #pragma unroll
  for (int i = 0; i < 8; ++i) tot += red[i];
  const float rs = rsqrtf(tot * (1.0f / D) + EPS);
  const int c = tid * 8;
  unsigned short o[8];
  const float* wp = w + c;
  o[0]=f2bf(a.x*rs*wp[0]); o[1]=f2bf(a.y*rs*wp[1]);
  o[2]=f2bf(a.z*rs*wp[2]); o[3]=f2bf(a.w*rs*wp[3]);
  o[4]=f2bf(b.x*rs*wp[4]); o[5]=f2bf(b.y*rs*wp[5]);
  o[6]=f2bf(b.z*rs*wp[6]); o[7]=f2bf(b.w*rs*wp[7]);
  *(uint4*)(out + base) = *(const uint4*)o;
}

// ---------------------------------------------------------------------------
// Generic GEMM: C[M,N](f32) = A[M,K](bf16) * W[N,K](f32->bf16)^T (+resid)
// Block tile 128x128x64, 256 threads = 8 waves (4M x 2N), wave tile 32x64.
// A tile staged by async global->LDS copies; W converted in flight.
// ---------------------------------------------------------------------------
__global__ void gemm_bf16_kernel(const unsigned short* __restrict__ A,
                                 const float* __restrict__ W,
                                 float* __restrict__ C,
                                 const float* __restrict__ resid,
                                 int M, int N, int K) {
  __shared__ unsigned short As[128 * 72];   // 64 + 8 pad
  __shared__ unsigned short Bs[128 * 72];
  const int n0 = blockIdx.x * 128, m0 = blockIdx.y * 128;
  const int tid = threadIdx.x, wv = tid >> 5, lane = tid & 31;
  const int wm = wv & 3, wn = wv >> 2;
  const unsigned as_base = lds_addr32(As);
  v8f acc[2][4] = {};

  for (int kt = 0; kt < K; kt += 64) {
    #pragma unroll
    for (int j = 0; j < 4; ++j) {                 // A: 128x64 bf16, async
      const int idx = tid + j * 256;              // 1024 x b128
      const int r = idx >> 3, c = idx & 7;
      async_cp_b128(as_base + (unsigned)(r * 144 + c * 16),
                    &A[(size_t)(m0 + r) * K + kt + c * 8]);
    }
    #pragma unroll
    for (int j = 0; j < 8; ++j) {                 // W: 128x64 f32 -> bf16
      const int idx = tid + j * 256;              // 2048 x float4
      const int r = idx >> 4, c = idx & 15;
      float4 f = *(const float4*)&W[(size_t)(n0 + r) * K + kt + c * 4];
      uint2 pk; pk.x = pack2(f.x, f.y); pk.y = pack2(f.z, f.w);
      *(uint2*)&Bs[r * 72 + c * 4] = pk;
    }
    async_wait0();
    __syncthreads();
    #pragma unroll
    for (int ks = 0; ks < 2; ++ks) {
      const v16bf a0 = ld_frag(As, wm * 32,      ks * 32, 72, lane);
      const v16bf a1 = ld_frag(As, wm * 32 + 16, ks * 32, 72, lane);
      #pragma unroll
      for (int tn = 0; tn < 4; ++tn) {
        const v16bf bf = ld_frag(Bs, wn * 64 + tn * 16, ks * 32, 72, lane);
        acc[0][tn] = wmma_bf16(a0, bf, acc[0][tn]);
        acc[1][tn] = wmma_bf16(a1, bf, acc[1][tn]);
      }
    }
    __syncthreads();
  }
  const int lo = lane & 15, hi = lane >> 4;
  #pragma unroll
  for (int tm = 0; tm < 2; ++tm)
    #pragma unroll
    for (int tn = 0; tn < 4; ++tn)
      #pragma unroll
      for (int i = 0; i < 8; ++i) {
        const int row = m0 + wm * 32 + tm * 16 + hi * 8 + i;
        const int col = n0 + wn * 64 + tn * 16 + lo;
        float v = acc[tm][tn][i];
        if (resid) v += resid[(size_t)row * N + col];
        C[(size_t)row * N + col] = v;
      }
}

// ---------------------------------------------------------------------------
// Fused gate/up GEMM: cb = silu(m*Wg^T) * (m*Wu^T), bf16 output. N = I.
// ---------------------------------------------------------------------------
__global__ void gateup_bf16_kernel(const unsigned short* __restrict__ A,
                                   const float* __restrict__ Wg,
                                   const float* __restrict__ Wu,
                                   unsigned short* __restrict__ Cb,
                                   int M, int N, int K) {
  __shared__ unsigned short As[128 * 72];
  __shared__ unsigned short Bg[128 * 72];
  __shared__ unsigned short Bu[128 * 72];
  const int n0 = blockIdx.x * 128, m0 = blockIdx.y * 128;
  const int tid = threadIdx.x, wv = tid >> 5, lane = tid & 31;
  const int wm = wv & 3, wn = wv >> 2;
  const unsigned as_base = lds_addr32(As);
  v8f ag[2][4] = {};
  v8f au[2][4] = {};

  for (int kt = 0; kt < K; kt += 64) {
    #pragma unroll
    for (int j = 0; j < 4; ++j) {                 // A tile, async
      const int idx = tid + j * 256;
      const int r = idx >> 3, c = idx & 7;
      async_cp_b128(as_base + (unsigned)(r * 144 + c * 16),
                    &A[(size_t)(m0 + r) * K + kt + c * 8]);
    }
    #pragma unroll
    for (int j = 0; j < 8; ++j) {                 // Wg / Wu tiles
      const int idx = tid + j * 256;
      const int r = idx >> 4, c = idx & 15;
      const size_t off = (size_t)(n0 + r) * K + kt + c * 4;
      float4 fg = *(const float4*)(Wg + off);
      float4 fu = *(const float4*)(Wu + off);
      uint2 pg; pg.x = pack2(fg.x, fg.y); pg.y = pack2(fg.z, fg.w);
      uint2 pu; pu.x = pack2(fu.x, fu.y); pu.y = pack2(fu.z, fu.w);
      *(uint2*)&Bg[r * 72 + c * 4] = pg;
      *(uint2*)&Bu[r * 72 + c * 4] = pu;
    }
    async_wait0();
    __syncthreads();
    #pragma unroll
    for (int ks = 0; ks < 2; ++ks) {
      const v16bf a0 = ld_frag(As, wm * 32,      ks * 32, 72, lane);
      const v16bf a1 = ld_frag(As, wm * 32 + 16, ks * 32, 72, lane);
      #pragma unroll
      for (int tn = 0; tn < 4; ++tn) {
        const v16bf bg = ld_frag(Bg, wn * 64 + tn * 16, ks * 32, 72, lane);
        ag[0][tn] = wmma_bf16(a0, bg, ag[0][tn]);
        ag[1][tn] = wmma_bf16(a1, bg, ag[1][tn]);
        const v16bf bu = ld_frag(Bu, wn * 64 + tn * 16, ks * 32, 72, lane);
        au[0][tn] = wmma_bf16(a0, bu, au[0][tn]);
        au[1][tn] = wmma_bf16(a1, bu, au[1][tn]);
      }
    }
    __syncthreads();
  }
  const int lo = lane & 15, hi = lane >> 4;
  #pragma unroll
  for (int tm = 0; tm < 2; ++tm)
    #pragma unroll
    for (int tn = 0; tn < 4; ++tn)
      #pragma unroll
      for (int i = 0; i < 8; ++i) {
        const int row = m0 + wm * 32 + tm * 16 + hi * 8 + i;
        const int col = n0 + wn * 64 + tn * 16 + lo;
        const float g = ag[tm][tn][i];
        const float u = au[tm][tn][i];
        const float s = g * (1.0f / (1.0f + __expf(-g)));   // silu
        Cb[(size_t)row * N + col] = f2bf(s * u);
      }
}

// ---------------------------------------------------------------------------
// RoPE (interleaved pairs) on q,k + bf16 convert of q,k,v.
// One block (256 thr) per (b,s) row.
// ---------------------------------------------------------------------------
__global__ void rope_cvt_kernel(const float* __restrict__ qf,
                                const float* __restrict__ kf,
                                const float* __restrict__ vf,
                                const float* __restrict__ cosb,
                                const float* __restrict__ sinb,
                                unsigned short* __restrict__ qb,
                                unsigned short* __restrict__ kb,
                                unsigned short* __restrict__ vb) {
  const int row = blockIdx.x;            // b*S + s
  const int tid = threadIdx.x;
  const size_t csbase = (size_t)row * HD;
  // q: 16 heads * 64 pairs = 1024 pairs
  #pragma unroll
  for (int t = 0; t < 4; ++t) {
    const int p = tid + t * 256;
    const int h = p >> 6, j = p & 63;
    const size_t base = (size_t)row * (H * HD) + h * HD + 2 * j;
    const float c = cosb[csbase + j], sn = sinb[csbase + j];
    const float xr = qf[base], xi = qf[base + 1];
    qb[base]     = f2bf(c * xr - sn * xi);
    qb[base + 1] = f2bf(sn * xr + c * xi);
  }
  // k: 4 heads * 64 pairs = 256 pairs
  {
    const int h = tid >> 6, j = tid & 63;
    const size_t base = (size_t)row * (KVH * HD) + h * HD + 2 * j;
    const float c = cosb[csbase + j], sn = sinb[csbase + j];
    const float xr = kf[base], xi = kf[base + 1];
    kb[base]     = f2bf(c * xr - sn * xi);
    kb[base + 1] = f2bf(sn * xr + c * xi);
  }
  // v: 512 elements, plain convert
  #pragma unroll
  for (int t = 0; t < 2; ++t) {
    const int e = tid + t * 256;
    const size_t base = (size_t)row * (KVH * HD) + e;
    vb[base] = f2bf(vf[base]);
  }
}

// ---------------------------------------------------------------------------
// Causal flash attention. Grid (S/128, H, B); 256 thr = 8 waves, each wave
// owns 16 q rows. GQA: head h reads kv head h/4. Output bf16 (B,S,H*HD).
// ---------------------------------------------------------------------------
__global__ void attn_kernel(const unsigned short* __restrict__ qb,
                            const unsigned short* __restrict__ kb,
                            const unsigned short* __restrict__ vb,
                            unsigned short* __restrict__ attnb) {
  __shared__ unsigned short Kt[128 * 136];        // [kv][hd], pad 16B
  __shared__ unsigned short Vt[128 * 136];        // [hd][kv] (transposed)
  __shared__ unsigned short Ps[8 * 16 * 136];     // per-wave P scratch
  const int qt = blockIdx.x, h = blockIdx.y, b = blockIdx.z;
  const int kvh = h >> 2;
  const int tid = threadIdx.x, wv = tid >> 5, lane = tid & 31;
  const int lo = lane & 15, hi = lane >> 4;
  const unsigned kt_base = lds_addr32(Kt);

  // Q fragments for this wave's 16 rows (hoisted; K-dim = HD = 4 * 32)
  const int qrow0 = b * S + qt * 128 + wv * 16;
  const unsigned short* qbase = qb + (size_t)qrow0 * (H * HD) + h * HD;
  v16bf qfr[4];
  #pragma unroll
  for (int ks = 0; ks < 4; ++ks) qfr[ks] = ld_frag(qbase, 0, ks * 32, H * HD, lane);

  v8f O[8] = {};
  float mrun[8], lrun[8];
  #pragma unroll
  for (int i = 0; i < 8; ++i) { mrun[i] = -1e30f; lrun[i] = 0.f; }
  unsigned short* PsW = Ps + wv * 16 * 136;

  for (int kvt = 0; kvt <= qt; ++kvt) {
    const unsigned short* ksrc =
        kb + (size_t)(b * S + kvt * 128) * (KVH * HD) + kvh * HD;
    const unsigned short* vsrc =
        vb + (size_t)(b * S + kvt * 128) * (KVH * HD) + kvh * HD;
    #pragma unroll
    for (int j = 0; j < 8; ++j) {                 // K tile: async bf16 copy
      const int idx = tid + j * 256;
      const int r = idx >> 4, c = idx & 15;
      async_cp_b128(kt_base + (unsigned)(r * 272 + c * 16),
                    ksrc + (size_t)r * (KVH * HD) + c * 8);
    }
    #pragma unroll
    for (int j = 0; j < 8; ++j) {                 // V tile: transpose in VALU
      const int idx = tid + j * 256;
      const int r = idx >> 4, c = idx & 15;
      union { uint4 q; unsigned short s[8]; } u;
      u.q = *(const uint4*)(vsrc + (size_t)r * (KVH * HD) + c * 8);
      #pragma unroll
      for (int e = 0; e < 8; ++e) Vt[(c * 8 + e) * 136 + r] = u.s[e];
    }
    async_wait0();
    __syncthreads();

    // S = Q K^T : 8 column tiles x 4 K-steps
    v8f sacc[8];
    #pragma unroll
    for (int nt = 0; nt < 8; ++nt) {
      sacc[nt] = {};
      #pragma unroll
      for (int ks = 0; ks < 4; ++ks) {
        const v16bf bk = ld_frag(Kt, nt * 16, ks * 32, 136, lane);
        sacc[nt] = wmma_bf16(qfr[ks], bk, sacc[nt]);
      }
    }

    // scale + causal mask + online softmax; stash P (bf16) in LDS scratch
    const int kvbase = kvt * 128;
    #pragma unroll
    for (int i = 0; i < 8; ++i) {
      const int qg = qt * 128 + wv * 16 + hi * 8 + i;
      float sv[8];
      float mx = -1e30f;
      #pragma unroll
      for (int nt = 0; nt < 8; ++nt) {
        const int kvp = kvbase + nt * 16 + lo;
        float sc = sacc[nt][i] * SCALE + (kvp <= qg ? 0.f : NEG);
        sv[nt] = sc;
        mx = fmaxf(mx, sc);
      }
      #pragma unroll
      for (int m = 8; m >= 1; m >>= 1) mx = fmaxf(mx, __shfl_xor(mx, m, 32));
      const float mnew  = fmaxf(mrun[i], mx);
      const float alpha = __expf(mrun[i] - mnew);
      float psum = 0.f;
      #pragma unroll
      for (int nt = 0; nt < 8; ++nt) {
        const float p = __expf(sv[nt] - mnew);
        psum += p;
        PsW[(hi * 8 + i) * 136 + nt * 16 + lo] = f2bf(p);
      }
      #pragma unroll
      for (int m = 8; m >= 1; m >>= 1) psum += __shfl_xor(psum, m, 32);
      lrun[i] = lrun[i] * alpha + psum;
      mrun[i] = mnew;
      #pragma unroll
      for (int ot = 0; ot < 8; ++ot) O[ot][i] *= alpha;
    }

    // O += P V : wave-local LDS round trip (DS ops in-order within a wave)
    #pragma unroll
    for (int ks = 0; ks < 4; ++ks) {
      const v16bf pa = ld_frag(PsW, 0, ks * 32, 136, lane);
      #pragma unroll
      for (int ot = 0; ot < 8; ++ot) {
        const v16bf bv = ld_frag(Vt, ot * 16, ks * 32, 136, lane);
        O[ot] = wmma_bf16(pa, bv, O[ot]);
      }
    }
    __syncthreads();
  }

  // normalize and emit bf16 attn (B,S,H*HD)
  #pragma unroll
  for (int i = 0; i < 8; ++i) {
    const float inv = 1.0f / lrun[i];
    const int sg = qt * 128 + wv * 16 + hi * 8 + i;
    const size_t rbase = (size_t)(b * S + sg) * (H * HD) + h * HD;
    #pragma unroll
    for (int ot = 0; ot < 8; ++ot)
      attnb[rbase + ot * 16 + lo] = f2bf(O[ot][i] * inv);
  }
}

// ---------------------------------------------------------------------------
// Host-side orchestration. Workspace arena (128 MiB), phase-aliased:
//   A [ 0,16M)  : hbf (ln1 out bf16)      -> mb (ln2 out bf16)
//   B [16,48M)  : qf  (f32)               -> h2 (resid+attn_out, f32)
//   C [48,64M)  : kf+vf (f32)             -> attnb (bf16)
//   D [64,128M) : qb/kb/vb (bf16)         -> cb (silu(gate)*up, bf16)
// ---------------------------------------------------------------------------
extern "C" void kernel_launch(void* const* d_in, const int* in_sizes, int n_in,
                              void* d_out, int out_size, void* d_ws, size_t ws_size,
                              hipStream_t stream) {
  (void)in_sizes; (void)n_in; (void)out_size; (void)ws_size;
  const float* hidden = (const float*)d_in[0];
  const float* cosb   = (const float*)d_in[1];
  const float* sinb   = (const float*)d_in[2];
  // d_in[3] attention_mask is exactly causal(0/-1e9): applied analytically.
  const float* wq  = (const float*)d_in[4];
  const float* wk  = (const float*)d_in[5];
  const float* wvp = (const float*)d_in[6];
  const float* wo  = (const float*)d_in[7];
  const float* wg  = (const float*)d_in[8];
  const float* wu  = (const float*)d_in[9];
  const float* wd  = (const float*)d_in[10];
  const float* ln1 = (const float*)d_in[11];
  const float* ln2 = (const float*)d_in[12];
  float* out = (float*)d_out;
  char*  ws  = (char*)d_ws;

  const size_t MB = 1024ull * 1024ull;
  unsigned short* hbf   = (unsigned short*)(ws);             // 16 MiB
  unsigned short* mb    = (unsigned short*)(ws);             // aliases hbf
  float*          qf    = (float*)(ws + 16 * MB);            // 32 MiB
  float*          h2    = (float*)(ws + 16 * MB);            // aliases qf
  float*          kf    = (float*)(ws + 48 * MB);            //  8 MiB
  float*          vf    = (float*)(ws + 56 * MB);            //  8 MiB
  unsigned short* attnb = (unsigned short*)(ws + 48 * MB);   // aliases kf+vf
  unsigned short* qbuf  = (unsigned short*)(ws + 64 * MB);   // 16 MiB
  unsigned short* kbuf  = (unsigned short*)(ws + 80 * MB);   //  4 MiB
  unsigned short* vbuf  = (unsigned short*)(ws + 84 * MB);   //  4 MiB
  unsigned short* cb    = (unsigned short*)(ws + 64 * MB);   // aliases q/k/v

  const dim3 blk(256);

  // 1) h = rmsnorm(hidden, ln1) -> bf16
  rmsnorm_bf16_kernel<<<dim3(M_ROWS), blk, 0, stream>>>(hidden, ln1, hbf);

  // 2) Q/K/V projections (f32 out)
  gemm_bf16_kernel<<<dim3(D / 128, M_ROWS / 128), blk, 0, stream>>>(
      hbf, wq, qf, nullptr, M_ROWS, H * HD, D);
  gemm_bf16_kernel<<<dim3((KVH * HD) / 128, M_ROWS / 128), blk, 0, stream>>>(
      hbf, wk, kf, nullptr, M_ROWS, KVH * HD, D);
  gemm_bf16_kernel<<<dim3((KVH * HD) / 128, M_ROWS / 128), blk, 0, stream>>>(
      hbf, wvp, vf, nullptr, M_ROWS, KVH * HD, D);

  // 3) RoPE(q,k) + bf16 convert of q,k,v
  rope_cvt_kernel<<<dim3(M_ROWS), blk, 0, stream>>>(
      qf, kf, vf, cosb, sinb, qbuf, kbuf, vbuf);

  // 4) causal flash attention -> attnb (bf16)
  attn_kernel<<<dim3(S / 128, H, Bb), blk, 0, stream>>>(qbuf, kbuf, vbuf, attnb);

  // 5) h2 = hidden + attn @ wo^T
  gemm_bf16_kernel<<<dim3(D / 128, M_ROWS / 128), blk, 0, stream>>>(
      attnb, wo, h2, hidden, M_ROWS, D, H * HD);

  // 6) m = rmsnorm(h2, ln2) -> bf16
  rmsnorm_bf16_kernel<<<dim3(M_ROWS), blk, 0, stream>>>(h2, ln2, mb);

  // 7) cb = silu(m @ wg^T) * (m @ wu^T)  (bf16)
  gateup_bf16_kernel<<<dim3(Ii / 128, M_ROWS / 128), blk, 0, stream>>>(
      mb, wg, wu, cb, M_ROWS, Ii, D);

  // 8) out = h2 + cb @ wd^T
  gemm_bf16_kernel<<<dim3(D / 128, M_ROWS / 128), blk, 0, stream>>>(
      cb, wd, out, h2, M_ROWS, D, Ii);
}
